// PatchMatch_13391708029996
// MI455X (gfx1250) — compile-verified, matrix operand order checked by reading, not run on hardware
//
#include <hip/hip_runtime.h>
#include <stdint.h>

#define H_  128
#define W_  128
#define HW_ 16384

typedef __attribute__((ext_vector_type(16))) _Float16 v16h;
typedef __attribute__((ext_vector_type(8)))  float    v8f;

// ---------------- threefry2x32 (JAX PRNG), host+device ----------------
__host__ __device__ __forceinline__ unsigned rotl32(unsigned v, int n){ return (v << n) | (v >> (32 - n)); }

__host__ __device__ inline void tf2x32(unsigned k0, unsigned k1, unsigned &x0, unsigned &x1){
  unsigned ks2 = k0 ^ k1 ^ 0x1BD11BDAu;
  x0 += k0; x1 += k1;
#define TFR(r) { x0 += x1; x1 = rotl32(x1, r); x1 ^= x0; }
  TFR(13) TFR(15) TFR(26) TFR(6)
  x0 += k1;  x1 += ks2 + 1u;
  TFR(17) TFR(29) TFR(16) TFR(24)
  x0 += ks2; x1 += k0 + 2u;
  TFR(13) TFR(15) TFR(26) TFR(6)
  x0 += k0;  x1 += k1 + 3u;
  TFR(17) TFR(29) TFR(16) TFR(24)
  x0 += k1;  x1 += ks2 + 4u;
  TFR(13) TFR(15) TFR(26) TFR(6)
  x0 += ks2; x1 += k0 + 5u;
#undef TFR
}

// JAX random_bits: counters [0..N) split into halves (x0 = i, x1 = i + N/2), then uniform = bitcast(1.0|mant)-1
__device__ __forceinline__ float jax_uniform(unsigned k0, unsigned k1, unsigned idx, unsigned halfN){
  unsigned j  = idx & (halfN - 1u);
  unsigned x0 = j, x1 = j + halfN;
  tf2x32(k0, k1, x0, x1);
  unsigned bits = (idx >= halfN) ? x1 : x0;
  return __uint_as_float(0x3f800000u | (bits >> 9)) - 1.0f;
}

// |x| via explicit VALU op: keeps the diff math scalar here and avoids the
// broken packed-f32 fabs lowering (bitcast/and idiom) in the gfx1250 backend.
__device__ __forceinline__ float absf_asm(float x){
  float r;
  asm("v_and_b32 %0, 0x7fffffff, %1" : "=v"(r) : "v"(x));
  return r;
}

// jnp.remainder(v, 128) -> [0, 128)
__device__ __forceinline__ float fixmod128(float v){
  float m = v - 128.0f * floorf(v * 0.0078125f);
  m = (m >= 128.0f) ? (m - 128.0f) : m;
  m = (m < 0.0f) ? 0.0f : m;
  return m;
}

__device__ __forceinline__ float ld_o(const float* o, int b, int y, int x){
  return o[(size_t)b * HW_ + ((y & 127) << 7) + (x & 127)];
}

// _propagate candidates (jnp.roll semantics: roll +1 reads from index-1, wrapped)
__device__ float prop_cand(const float* o, int b, int y, int x, int k){
  switch (k){
    case 0:  return ld_o(o,b,y,x);
    case 1:  return ld_o(o,b,y,x-1);
    case 2:  return ld_o(o,b,y,x+1);
    case 3:  return ld_o(o,b,y-1,x);
    case 4:  return ld_o(o,b,y+1,x);
    case 5:  return 2.f*ld_o(o,b,y,x-1)   - ld_o(o,b,y,x-2);
    case 6:  return 2.f*ld_o(o,b,y,x+1)   - ld_o(o,b,y,x+2);
    case 7:  return 2.f*ld_o(o,b,y-1,x)   - ld_o(o,b,y-2,x);
    case 8:  return 2.f*ld_o(o,b,y+1,x)   - ld_o(o,b,y+2,x);
    case 9:  return 2.f*ld_o(o,b,y-1,x-1) - ld_o(o,b,y-2,x-2);
    case 10: return 2.f*ld_o(o,b,y+1,x+1) - ld_o(o,b,y+2,x+2);
    case 11: return 2.f*ld_o(o,b,y-1,x+1) - ld_o(o,b,y-2,x+2);
    default: return 2.f*ld_o(o,b,y+1,x-1) - ld_o(o,b,y+2,x-2);
  }
}

// init: ox = 128*u1 - lx, oy = 128*u2 - ly  (the initial _fix is identity here)
__global__ __launch_bounds__(256) void pm_init(float* ox, float* oy,
                                               unsigned a0, unsigned a1,
                                               unsigned b0, unsigned b1){
  unsigned i = blockIdx.x * 256u + threadIdx.x;   // < 131072
  int x = (int)(i & 127u), y = (int)((i >> 7) & 127u);
  float u1 = jax_uniform(a0, a1, i, 65536u);
  float u2 = jax_uniform(b0, b1, i, 65536u);
  ox[i] = 128.0f * u1 - (float)x;
  oy[i] = 128.0f * u2 - (float)y;
}

// ---------------- fused evaluate ----------------
// One block per (row y, batch b). 8 waves x 16 pixels = one full row.
// Wave layout (WMMA column striping; contiguous K order is this code's choice):
//   lane&15  = pixel (B-matrix column / D column)
//   lane>>4  = which K-half of the 32 intra-group channels this lane carries (B rows K = hi*16+i)
// Nine accumulating v_wmma_f32_16x16x32_f16: A = one-hot row selector for pair p (lane&15==p),
// B = |center - warped| per (channel, pixel)  =>  D[p][pixel] = cross-group sum for pair p.
template<int GS, bool SAMPLE>
__global__ __launch_bounds__(256) void pm_eval(
    const float* __restrict__ feat,
    const float* ox_in, const float* oy_in,
    float* ox_out, float* oy_out,
    unsigned nk0, unsigned nk1)
{
  constexpr int C = 3 * GS;
  __shared__ float fsm[C * W_];

  const int tid = threadIdx.x;
  const int y   = blockIdx.x;
  const int b   = blockIdx.y;
  const float* fb = feat + (size_t)b * (size_t)(C * HW_);

  // stage this row's center features (reused by every candidate)
  for (int idx = tid; idx < C * W_; idx += 256){
    int c = idx >> 7, x = idx & 127;
    fsm[idx] = fb[(size_t)c * HW_ + (y << 7) + x];
  }
  __syncthreads();

  const int lane = tid & 31;
  const int wv   = tid >> 5;
  const int px   = (wv << 4) + (lane & 15);
  const int hi   = lane >> 4;
  const float lx = (float)px, lyf = (float)y;

  // center features for this lane's channel subset (B rows K = hi*16 + i)
  float fcc[3][16];
#pragma unroll
  for (int g = 0; g < 3; ++g)
#pragma unroll
    for (int i = 0; i < 16; ++i){
      int kk = hi * 16 + i;
      fcc[g][i] = (kk < GS) ? fsm[(g * GS + kk) * W_ + px] : 0.0f;
    }

  float sx0 = 0.f, sx1 = 0.f, sy0 = 0.f, sy1 = 0.f, cox = 0.f, coy = 0.f;
  if (SAMPLE){
    size_t o = (size_t)b * HW_ + (y << 7) + px;
    cox = ox_in[o]; coy = oy_in[o];
    unsigned i0 = ((unsigned)(b * 2 + 0) << 14) + ((unsigned)y << 7) + (unsigned)px;
    unsigned i1 = ((unsigned)(b * 2 + 1) << 14) + ((unsigned)y << 7) + (unsigned)px;
    float n0 = jax_uniform(nk0, nk1, i0, 131072u);
    float n1 = jax_uniform(nk0, nk1, i1, 131072u);
    sx0 = cox - 4.0f + 4.0f * n0;   // min + 4*(0+n0)
    sx1 = cox + 4.0f * n1;          // min + 4*(1+n1)
    sy0 = coy - 4.0f + 4.0f * n0;
    sy1 = coy + 4.0f * n1;
  }

  float M = -1e30f, den = 0.f, axs = 0.f, ays = 0.f;
  const int NK = SAMPLE ? 5 : 13;

#pragma unroll 1
  for (int k = 0; k < NK; ++k){
    float pox, poy;
    if (SAMPLE){
      pox = (k == 4) ? cox : ((k & 1) ? sx1 : sx0);
      poy = (k == 4) ? coy : ((k & 2) ? sy1 : sy0);
    } else {
      pox = prop_cand(ox_in, b, y, px, k);
      poy = prop_cand(oy_in, b, y, px, k);
    }
    // _fix: wrap absolute target into [0,128)
    float tx = fixmod128(pox + lx);  pox = tx - lx;
    float ty = fixmod128(poy + lyf); poy = ty - lyf;
    // _evaluate clip + bilinear setup
    float rx = fminf(tx, 127.0f), ry = fminf(ty, 127.0f);
    float fx0 = floorf(rx), fy0 = floorf(ry);
    float wxf = rx - fx0, wyf = ry - fy0;
    int x0 = (int)fx0, y0 = (int)fy0;
    int x1 = (x0 < 127) ? x0 + 1 : 127;
    int y1 = (y0 < 127) ? y0 + 1 : 127;
    float w00 = (1.f - wxf) * (1.f - wyf), w01 = wxf * (1.f - wyf);
    float w10 = (1.f - wxf) * wyf,         w11 = wxf * wyf;
    int o00 = (y0 << 7) + x0, o01 = (y0 << 7) + x1;
    int o10 = (y1 << 7) + x0, o11 = (y1 << 7) + x1;

    // bilinear warp of this lane's channel subset (gathers are L2-resident)
    float wrp[3][16];
#pragma unroll
    for (int g = 0; g < 3; ++g)
#pragma unroll
      for (int i = 0; i < 16; ++i){
        int kk = hi * 16 + i;
        float v = 0.0f;
        if (kk < GS){
          const float* cp = fb + (size_t)(g * GS + kk) * HW_;
          v = w00 * cp[o00] + w01 * cp[o01] + w10 * cp[o10] + w11 * cp[o11];
        }
        wrp[g][i] = v;
      }

    // nine 16x16x32 WMMAs: D[p][pixel] = sum_k |f_g1[k] - w_g2[k]|
    v8f acc;
#pragma unroll
    for (int r = 0; r < 8; ++r) acc[r] = 0.0f;
#pragma unroll
    for (int g1 = 0; g1 < 3; ++g1){
#pragma unroll
      for (int g2 = 0; g2 < 3; ++g2){
        const int p = g1 * 3 + g2;
        _Float16 sel = ((lane & 15) == p) ? (_Float16)1.0f : (_Float16)0.0f;
        v16h av, bv;
#pragma unroll
        for (int i = 0; i < 16; ++i){
          av[i] = sel;                                                  // A: one-hot row p
          bv[i] = (_Float16)absf_asm(fcc[g1][i] - wrp[g2][i]);          // B: |diff| (K x pixel)
        }
        acc = __builtin_amdgcn_wmma_f32_16x16x32_f16(false, av, false, bv,
                                                     (short)0, acc, false, false);
      }
    }

    // pairs 0..7 live in-lane (lanes 0-15); pair 8 lives in partner lane (+16), VGPR0
    float p8 = __shfl(acc[0], (lane & 15) + 16, 32);
    float smin = fminf(acc[0], acc[1]);
    smin = fminf(smin, fminf(acc[2], acc[3]));
    smin = fminf(smin, fminf(acc[4], acc[5]));
    smin = fminf(smin, fminf(acc[6], acc[7]));
    smin = fminf(smin, p8);
    float s = -smin * (1.0f / (float)GS);       // max over 9 of mean(-|diff|)

    // online softmax (temp 1000)
    float newM = fmaxf(M, s);
    float rsc  = __expf(1000.0f * (M - newM));
    float e    = __expf(1000.0f * (s - newM));
    den = den * rsc + e;
    axs = axs * rsc + e * pox;
    ays = ays * rsc + e * poy;
    M = newM;
  }

  if (lane < 16){
    float wx = axs / den, wy = ays / den;
    size_t o = (size_t)b * HW_ + (y << 7) + px;
    ox_out[o] = fminf(fmaxf(wx + lx,  0.0f), 127.0f) - lx;
    oy_out[o] = fminf(fmaxf(wy + lyf, 0.0f), 127.0f) - lyf;
  }
}

extern "C" void kernel_launch(void* const* d_in, const int* in_sizes, int n_in,
                              void* d_out, int out_size, void* d_ws, size_t ws_size,
                              hipStream_t stream)
{
  (void)in_sizes; (void)n_in; (void)out_size; (void)ws_size;
  const float* ZM   = (const float*)d_in[0];
  const float* CNN  = (const float*)d_in[1];
  const float* CNNg = (const float*)d_in[2];

  float* oxA = (float*)d_ws;          // buffer A in workspace (1 MB)
  float* oyA = oxA + 131072;
  float* oxB = (float*)d_out;         // buffer B = output buffer
  float* oyB = oxB + 131072;

  // subkeys of jax.random.split(key(42), 4) computed on host (threefry2x32)
  unsigned o0[4], o1[4];
  for (int j = 0; j < 4; ++j){
    unsigned a = (unsigned)j, c = (unsigned)j + 4u;
    tf2x32(0u, 42u, a, c);
    o0[j] = a; o1[j] = c;
  }
  // k1 = {o0[0],o0[1]}, k2 = {o0[2],o0[3]}, ks[0] = {o1[0],o1[1]}, ks[1] = {o1[2],o1[3]}

  pm_init<<<512, 256, 0, stream>>>(oxA, oyA, o0[0], o0[1], o0[2], o0[3]);

  dim3 grid(H_, 8);
  // ZM iterations (C=36, group 12)
  pm_eval<12, false><<<grid, 256, 0, stream>>>(ZM,   oxA, oyA, oxB, oyB, 0u, 0u);
  pm_eval<12, false><<<grid, 256, 0, stream>>>(ZM,   oxB, oyB, oxA, oyA, 0u, 0u);
  // CNN iteration 0
  pm_eval<32, false><<<grid, 256, 0, stream>>>(CNN,  oxA, oyA, oxB, oyB, 0u, 0u);
  pm_eval<32, true ><<<grid, 256, 0, stream>>>(CNNg, oxB, oyB, oxA, oyA, o1[0], o1[1]);
  // CNN iteration 1 (final sample-eval reads/writes d_out in place: per-pixel only)
  pm_eval<32, false><<<grid, 256, 0, stream>>>(CNN,  oxA, oyA, oxB, oyB, 0u, 0u);
  pm_eval<32, true ><<<grid, 256, 0, stream>>>(CNNg, oxB, oyB, oxB, oyB, o1[2], o1[3]);
}
